// MultiheadSelfAttentionALiBi_15367392985690
// MI455X (gfx1250) — matrix operand order repaired and checked by reading, not compiled, not run on hardware
//
#include <hip/hip_runtime.h>

// Problem constants (match reference)
#define T_SEQ 2048
#define NB    2
#define NH    16
#define HD    64
#define DM    1024

typedef __attribute__((ext_vector_type(16))) __bf16 bf16x16;
typedef __attribute__((ext_vector_type(8)))  float  f32x8;
typedef unsigned short u16;

union Frag {            // 16 bf16 = 8 VGPRs, loadable as two b128
  bf16x16 v;
  uint4   q[2];
};
union AccF {            // 8 f32 accumulator
  f32x8 v;
  float f[8];
};

__device__ __forceinline__ u16 f2bf(float x) {
  __bf16 h = (__bf16)x;                // RNE convert
  return __builtin_bit_cast(u16, h);
}

__device__ __forceinline__ f32x8 wmma_bf16(const Frag& a, const Frag& b, f32x8 c) {
  // 8 args: (neg_a, A, neg_b, B, c_mod, C, reuse_a, reuse_b)
  return __builtin_amdgcn_wmma_f32_16x16x32_bf16(false, a.v, false, b.v,
                                                 (short)0, c, false, false);
}

// ---------------------------------------------------------------------------
// One-time converts: f32 -> bf16 (straight) and f32 -> bf16 transposed.
// These run once per launch; the converted operands then live in L2 (192 MB)
// across all their reuses, so the GEMMs can be LDS-free.
// ---------------------------------------------------------------------------
__global__ void __launch_bounds__(256)
convert_f32_bf16(const float* __restrict__ src, u16* __restrict__ dst, int n)
{
  int i = (blockIdx.x * 256 + threadIdx.x) * 8;
  if (i >= n) return;
  float4 a = *(const float4*)(src + i);
  float4 b = *(const float4*)(src + i + 4);
  union { u16 h[8]; uint4 q; } u;
  u.h[0] = f2bf(a.x); u.h[1] = f2bf(a.y); u.h[2] = f2bf(a.z); u.h[3] = f2bf(a.w);
  u.h[4] = f2bf(b.x); u.h[5] = f2bf(b.y); u.h[6] = f2bf(b.z); u.h[7] = f2bf(b.w);
  *(uint4*)(dst + i) = u.q;
}

// src[R][C] f32 -> dst[C][R] bf16, 32x32 LDS tile (coalesced both sides)
__global__ void __launch_bounds__(256)
transpose_convert_bf16(const float* __restrict__ src, u16* __restrict__ dst,
                       int R, int C)
{
  __shared__ u16 tile[32][33];
  const int c0 = blockIdx.x * 32;
  const int r0 = blockIdx.y * 32;
  const int tr = threadIdx.x >> 3;        // 0..31
  const int tc = (threadIdx.x & 7) * 4;   // 0,4,...,28

  float4 v = *(const float4*)(src + (size_t)(r0 + tr) * C + c0 + tc);
  tile[tr][tc + 0] = f2bf(v.x);
  tile[tr][tc + 1] = f2bf(v.y);
  tile[tr][tc + 2] = f2bf(v.z);
  tile[tr][tc + 3] = f2bf(v.w);
  __syncthreads();

  uint2 pk;
  pk.x = (unsigned)tile[tc + 0][tr] | ((unsigned)tile[tc + 1][tr] << 16);
  pk.y = (unsigned)tile[tc + 2][tr] | ((unsigned)tile[tc + 3][tr] << 16);
  *(uint2*)(dst + (size_t)(c0 + tr) * R + r0 + tc) = pk;
}

// ---------------------------------------------------------------------------
// LDS-free bf16 WMMA GEMM: C[M,N] = A[M,K] (bf16 [m][k]) x B (bf16 [n][k]) + bias
// Workgroup 256 threads = 8 waves (2 x 4); wave tile 32(M) x 64(N):
// 2x4 = 8 WMMA accumulators, 12 b128 fragment loads per 32-k step.
// MODE 0: epilogue scatters bf16 into Q[bh,t,d] / K[bh,t,d] / Vt[bh,d,t]
// MODE 1: epilogue stores f32 to Cout[M,N]
// ---------------------------------------------------------------------------
template<int MODE>
__global__ void __launch_bounds__(256)
gemm_bf16_wmma(const u16* __restrict__ Abf, const u16* __restrict__ Bt,
               const float* __restrict__ bias, int M, int N, int K,
               u16* __restrict__ Qbf, u16* __restrict__ Kbf, u16* __restrict__ Vt,
               float* __restrict__ Cout)
{
  const int lane = threadIdx.x & 31;
  const int wave = threadIdx.x >> 5;
  const int wm   = wave & 1;      // 2 wave rows
  const int wn   = wave >> 1;     // 4 wave cols
  const int kh   = lane >> 4;     // lane half (wave32)
  const int lc   = lane & 15;

  const int row0 = blockIdx.x * 64;
  const int col0 = blockIdx.y * 256;

  AccF acc[2][4];
#pragma unroll
  for (int mi = 0; mi < 2; ++mi)
#pragma unroll
    for (int ni = 0; ni < 4; ++ni)
#pragma unroll
      for (int r = 0; r < 8; ++r) acc[mi][ni].f[r] = 0.f;

  // per-lane fragment base pointers (A layout rows / B layout cols)
  const u16* abase[2];
#pragma unroll
  for (int mi = 0; mi < 2; ++mi)
    abase[mi] = Abf + (size_t)(row0 + wm * 32 + mi * 16 + lc) * K;
  const u16* bbase[4];
#pragma unroll
  for (int ni = 0; ni < 4; ++ni)
    bbase[ni] = Bt + (size_t)(col0 + wn * 64 + ni * 16 + lc) * K;

  for (int kb = 0; kb < K; kb += 32) {
    Frag af[2], bfr[4];
#pragma unroll
    for (int mi = 0; mi < 2; ++mi) {
      const u16* p = abase[mi] + kb;
      af[mi].q[0] = *(const uint4*)(p + kh * 8);        // k in [kh*8, +8)
      af[mi].q[1] = *(const uint4*)(p + 16 + kh * 8);   // k in [16+kh*8, +8)
    }
#pragma unroll
    for (int ni = 0; ni < 4; ++ni) {
      const u16* p = bbase[ni] + kb;
      bfr[ni].q[0] = *(const uint4*)(p + kh * 8);       // k in [kh*8, +8)
      bfr[ni].q[1] = *(const uint4*)(p + 16 + kh * 8);
    }
    if (kb + 32 < K) {   // global_prefetch_b8 of next k-step
      __builtin_prefetch(abase[0] + kb + 32, 0, 1);
      __builtin_prefetch(bbase[0] + kb + 32, 0, 1);
    }
#pragma unroll
    for (int mi = 0; mi < 2; ++mi)
#pragma unroll
      for (int ni = 0; ni < 4; ++ni)
        acc[mi][ni].v = wmma_bf16(af[mi], bfr[ni], acc[mi][ni].v);
  }

  // ---- epilogue ----
#pragma unroll
  for (int mi = 0; mi < 2; ++mi)
#pragma unroll
    for (int ni = 0; ni < 4; ++ni)
#pragma unroll
      for (int r = 0; r < 8; ++r) {
        int row = row0 + wm * 32 + mi * 16 + kh * 8 + r;   // C layout: m = r + 8*half
        int col = col0 + wn * 64 + ni * 16 + lc;           // n = lane&15
        float v = acc[mi][ni].f[r] + bias[col];
        if (MODE == 0) {
          int which = col >> 10;          // 0=q 1=k 2=v
          int rem   = col & 1023;
          int h     = rem >> 6;
          int d     = rem & 63;
          int b     = row >> 11;          // / T_SEQ
          int t     = row & 2047;
          size_t bh = (size_t)(b * NH + h);
          u16 bv = f2bf(v);
          if (which == 0)      Qbf[(bh * T_SEQ + t) * HD + d] = bv;
          else if (which == 1) Kbf[(bh * T_SEQ + t) * HD + d] = bv;
          else                 Vt [(bh * HD + d) * T_SEQ + t] = bv;   // transposed
        } else {
          Cout[(size_t)row * N + col] = v;
        }
      }
}

// ---------------------------------------------------------------------------
// Flash attention with ALiBi + key padding mask, bf16 WMMA, wave32.
// Block = 128 threads = 4 waves; each wave owns a 16-row q tile (64 q rows/WG).
// 32-key chunks: 4 WMMAs for S = Q K^T, 4 WMMAs for O += P V.
// Softmax in base-2 domain (exp2f -> v_exp_f32); ALiBi split into a
// loop-invariant per-row term + a running per-chunk key term (no cvt in loop).
// Output written directly as bf16 (feeds GEMM2 without conversion).
// ---------------------------------------------------------------------------
__global__ void __launch_bounds__(128)
attn_alibi_flash(const u16* __restrict__ Qbf, const u16* __restrict__ Kbf,
                 const u16* __restrict__ Vt,  const int* __restrict__ amask,
                 const float* __restrict__ slopes, u16* __restrict__ attn_bf)
{
  __shared__ u16 sP[4][16 * 32];   // per-wave P tile [16 rows][32 keys]

  const int lane = threadIdx.x & 31;
  const int wave = threadIdx.x >> 5;
  const int kh   = lane >> 4;
  const int lc   = lane & 15;
  const int bh   = blockIdx.y;          // b*NH + h
  const int b    = bh >> 4;
  const int h    = bh & 15;
  const int q0   = blockIdx.x * 64 + wave * 16;

  const float LOG2E = 1.44269504088896f;
  const float sc2  = 0.125f * LOG2E;         // 1/sqrt(64) in base-2 domain
  const float sl2  = slopes[h] * LOG2E;
  const float sl16 = sl2 * 16.f;
  const float sl32 = sl2 * 32.f;

  // loop-invariant ALiBi row terms: sl2 * (lc - qi)
  float rowA[8];
#pragma unroll
  for (int r = 0; r < 8; ++r)
    rowA[r] = sl2 * (float)(lc - (q0 + kh * 8 + r));

  // Q fragments (A layout), loaded once: row = q0 + lc
  const u16* qrow = Qbf + ((size_t)bh * T_SEQ + q0 + lc) * HD;
  Frag qf[2];
#pragma unroll
  for (int ks = 0; ks < 2; ++ks) {
    qf[ks].q[0] = *(const uint4*)(qrow + ks * 32 + kh * 8);
    qf[ks].q[1] = *(const uint4*)(qrow + ks * 32 + 16 + kh * 8);
  }

  AccF oacc[4];
#pragma unroll
  for (int dt = 0; dt < 4; ++dt)
#pragma unroll
    for (int r = 0; r < 8; ++r) oacc[dt].f[r] = 0.f;

  float m_run[8], l_run[8];
#pragma unroll
  for (int r = 0; r < 8; ++r) { m_run[r] = -3.0e38f; l_run[r] = 0.f; }

  const int* mrow = amask + b * T_SEQ;
  u16* pw = sP[wave];
  float keyA = 0.f;                       // sl2 * key0, running

  for (int key0 = 0; key0 < T_SEQ; key0 += 32, keyA += sl32) {
    // ---- S = Q K^T for 32 keys (two 16x16 tiles) ----
    AccF s[2];
#pragma unroll
    for (int nt = 0; nt < 2; ++nt) {
#pragma unroll
      for (int r = 0; r < 8; ++r) s[nt].f[r] = 0.f;
      const u16* krow = Kbf + ((size_t)bh * T_SEQ + key0 + nt * 16 + lc) * HD;
#pragma unroll
      for (int ks = 0; ks < 2; ++ks) {
        Frag kf;                                       // B layout: d in [kh*8,+8) and [16+kh*8,+8)
        kf.q[0] = *(const uint4*)(krow + ks * 32 + kh * 8);
        kf.q[1] = *(const uint4*)(krow + ks * 32 + 16 + kh * 8);
        s[nt].v = wmma_bf16(qf[ks], kf, s[nt].v);
      }
    }

    const bool msk0 = (mrow[key0 + lc] == 0);
    const bool msk1 = (mrow[key0 + 16 + lc] == 0);
    const float a0  = keyA;               // ALiBi key terms for the two tiles
    const float a1  = keyA + sl16;

    // ---- online softmax per row (rows r+8*kh live in VGPR r of this half) ----
#pragma unroll
    for (int r = 0; r < 8; ++r) {
      float s0 = msk0 ? -__builtin_inff() : (s[0].f[r] * sc2 + a0 + rowA[r]);
      float s1 = msk1 ? -__builtin_inff() : (s[1].f[r] * sc2 + a1 + rowA[r]);
      float rm = fmaxf(s0, s1);
#pragma unroll
      for (int off = 1; off < 16; off <<= 1)
        rm = fmaxf(rm, __shfl_xor(rm, off, 32));       // reduce over 16-lane half
      float mnew  = fmaxf(m_run[r], rm);
      float alpha = exp2f(m_run[r] - mnew);
      float p0 = exp2f(s0 - mnew);
      float p1 = exp2f(s1 - mnew);
      float rs = p0 + p1;
#pragma unroll
      for (int off = 1; off < 16; off <<= 1)
        rs += __shfl_xor(rs, off, 32);
      l_run[r] = l_run[r] * alpha + rs;
      m_run[r] = mnew;
#pragma unroll
      for (int dt = 0; dt < 4; ++dt) oacc[dt].f[r] *= alpha;

      const int prow = (kh * 8 + r) * 32;              // P[16][32] in LDS
      pw[prow + lc]      = f2bf(p0);
      pw[prow + 16 + lc] = f2bf(p1);
    }

    // ---- re-read P in A layout (DS ops are in-order within a wave) ----
    Frag pf;
    const u16* pr = pw + lc * 32;
    pf.q[0] = *(const uint4*)(pr + kh * 16);
    pf.q[1] = *(const uint4*)(pr + kh * 16 + 8);

    // ---- O += P V : V^T[bh][d][t] gives contiguous-k B fragments ----
#pragma unroll
    for (int dt = 0; dt < 4; ++dt) {
      const u16* vp = Vt + ((size_t)bh * HD + dt * 16 + lc) * T_SEQ + key0 + kh * 16;
      Frag vf;
      vf.q[0] = *(const uint4*)vp;
      vf.q[1] = *(const uint4*)(vp + 8);
      oacc[dt].v = wmma_bf16(pf, vf, oacc[dt].v);
    }
  }

  // ---- normalize + write bf16 attn output [B*T][DM] ----
#pragma unroll
  for (int r = 0; r < 8; ++r) {
    float inv = 1.0f / l_run[r];
    int   t   = q0 + kh * 8 + r;
    u16* orow = attn_bf + ((size_t)b * T_SEQ + t) * DM + h * HD;
#pragma unroll
    for (int dt = 0; dt < 4; ++dt)
      orow[dt * 16 + lc] = f2bf(oacc[dt].f[r] * inv);
  }
}

// ---------------------------------------------------------------------------
extern "C" void kernel_launch(void* const* d_in, const int* in_sizes, int n_in,
                              void* d_out, int out_size, void* d_ws, size_t ws_size,
                              hipStream_t stream) {
  const float* x      = (const float*)d_in[0];
  const int*   amask  = (const int*)  d_in[1];
  const float* W_qkv  = (const float*)d_in[2];
  const float* b_qkv  = (const float*)d_in[3];
  const float* W_out  = (const float*)d_in[4];
  const float* b_out  = (const float*)d_in[5];
  const float* slopes = (const float*)d_in[6];
  float* out = (float*)d_out;

  const int M = NB * T_SEQ;                               // 4096
  const size_t xElems  = (size_t)M * DM;                  // 4M
  const size_t qkElems = (size_t)NB * NH * T_SEQ * HD;    // 4M bf16 each

  u16* Xbf    = (u16*)d_ws;                 // [4096][1024]
  u16* Wqkvt  = Xbf   + xElems;             // [3072][1024]
  u16* Woutt  = Wqkvt + (size_t)3 * DM * DM;// [1024][1024]
  u16* Qbf    = Woutt + (size_t)DM * DM;
  u16* Kbf    = Qbf + qkElems;
  u16* Vt     = Kbf + qkElems;
  u16* attnbf = Vt  + qkElems;              // [4096][1024]

  // 0) one-time bf16 conversions (operands then live in L2 across reuses)
  convert_f32_bf16<<<dim3((unsigned)(xElems / 2048)), dim3(256), 0, stream>>>(
      x, Xbf, (int)xElems);
  transpose_convert_bf16<<<dim3(3 * DM / 32, DM / 32), dim3(256), 0, stream>>>(
      W_qkv, Wqkvt, DM, 3 * DM);
  transpose_convert_bf16<<<dim3(DM / 32, DM / 32), dim3(256), 0, stream>>>(
      W_out, Woutt, DM, DM);

  // 1) QKV projection, scatter bf16 Q / K / V^T
  dim3 g1(M / 64, (3 * DM) / 256);
  gemm_bf16_wmma<0><<<g1, dim3(256), 0, stream>>>(
      Xbf, Wqkvt, b_qkv, M, 3 * DM, DM, Qbf, Kbf, Vt, nullptr);

  // 2) flash attention with ALiBi + key mask -> bf16
  dim3 g2(T_SEQ / 64, NB * NH);
  attn_alibi_flash<<<g2, dim3(128), 0, stream>>>(Qbf, Kbf, Vt, amask, slopes, attnbf);

  // 3) output projection -> d_out (f32)
  dim3 g3(M / 64, DM / 256);
  gemm_bf16_wmma<1><<<g3, dim3(256), 0, stream>>>(
      attnbf, Woutt, b_out, M, DM, DM, nullptr, nullptr, nullptr, out);
}
